// MambaConcatFusion_40209483825878
// MI455X (gfx1250) — matrix operand order repaired and checked by reading, not compiled, not run on hardware
//
#include <hip/hip_runtime.h>
#include <math.h>

typedef __bf16 bf16;
typedef __attribute__((ext_vector_type(16))) __bf16 v16bf;
typedef __attribute__((ext_vector_type(8)))  __bf16 v8bf;
typedef __attribute__((ext_vector_type(8)))  float  v8f;

union bf16x16 { v16bf v; v8bf h[2]; };

static constexpr int Bsz  = 8;
static constexpr int Lseq = 512;
static constexpr int H    = 768;
static constexpr int DIN  = 1536;
static constexpr int Nst  = 64;
static constexpr int Rrk  = 48;
static constexpr int RrkPad = 64;               // dt GEMM K padded to 64
static constexpr int Kcv  = 4;
static constexpr int Mrows = Bsz * Lseq;        // 4096
static constexpr int NDBL  = Rrk + 2 * Nst;     // 176 logical
static constexpr int NDBLP = 192;               // padded to multiple of 64

// ---------------------------------------------------------------------------
// Fragment loads matching the CDNA5 WMMA register layouts (all 16B-aligned).
// A (16x32 bf16): lane half selects K+0 / K+8; two contiguous 8-elem runs.
// B (32x16 bf16, from transposed weights): same two 8-elem runs as A (hi-lanes K+8).
// ---------------------------------------------------------------------------
__device__ __forceinline__ v16bf load_fragA(const bf16* __restrict__ p, int kbA)
{
    bf16x16 f;
    f.h[0] = *(const v8bf*)(p + kbA);
    f.h[1] = *(const v8bf*)(p + 16 + kbA);
    return f.v;
}
__device__ __forceinline__ v16bf load_fragB(const bf16* __restrict__ p, int kbB)
{
    bf16x16 f;
    f.h[0] = *(const v8bf*)(p + kbB);
    f.h[1] = *(const v8bf*)(p + 16 + kbB);
    return f.v;
}

struct Frags { v16bf a0, a1, b0, b1, b2, b3; };

// ---------------------------------------------------------------------------
// bf16 WMMA GEMM:  out[M,N] = A[M,K] @ W[K,N], W supplied transposed Wt[N,K].
// Block = 256 threads = 8 waves stacked along M; wave tile = 32(M) x 64(N)
// (8 accumulators).  Requires M%256==0, N%64==0, K%64==0 (K double-buffered).
// ---------------------------------------------------------------------------
template <bool HAS_BIAS, bool SOFTPLUS, bool OUT_BF16>
__global__ __launch_bounds__(256) void gemm_wmma_bf16(
    const bf16* __restrict__ A, const bf16* __restrict__ Wt,
    const float* __restrict__ bias,
    float* __restrict__ outF, bf16* __restrict__ outB,
    int M, int N, int K)
{
    const int lane   = threadIdx.x & 31;
    const int wave   = threadIdx.x >> 5;
    const int tileM  = blockIdx.y * 256 + wave * 32;
    const int tileN0 = blockIdx.x * 64;
    const int half   = lane >> 4;        // 0 | 1
    const int mn     = lane & 15;
    const int kbA    = half * 8;         // A: hi-lanes hold K+8
    const int kbB    = half * 8;         // B: hi-lanes hold K+8

    const bf16* __restrict__ Ar0 = A + (size_t)(tileM + mn) * K;
    const bf16* __restrict__ Ar1 = A + (size_t)(tileM + 16 + mn) * K;
    const bf16* __restrict__ Br0 = Wt + (size_t)(tileN0 +  0 + mn) * K;
    const bf16* __restrict__ Br1 = Wt + (size_t)(tileN0 + 16 + mn) * K;
    const bf16* __restrict__ Br2 = Wt + (size_t)(tileN0 + 32 + mn) * K;
    const bf16* __restrict__ Br3 = Wt + (size_t)(tileN0 + 48 + mn) * K;

    v8f acc00 = {0.f,0.f,0.f,0.f,0.f,0.f,0.f,0.f};
    v8f acc01 = acc00, acc02 = acc00, acc03 = acc00;
    v8f acc10 = acc00, acc11 = acc00, acc12 = acc00, acc13 = acc00;

    auto load_step = [&](Frags& f, int k0) {
        f.a0 = load_fragA(Ar0 + k0, kbA);
        f.a1 = load_fragA(Ar1 + k0, kbA);
        f.b0 = load_fragB(Br0 + k0, kbB);
        f.b1 = load_fragB(Br1 + k0, kbB);
        f.b2 = load_fragB(Br2 + k0, kbB);
        f.b3 = load_fragB(Br3 + k0, kbB);
    };
    auto mma_step = [&](const Frags& f) {
        acc00 = __builtin_amdgcn_wmma_f32_16x16x32_bf16(false, f.a0, false, f.b0, (short)0, acc00, false, false);
        acc01 = __builtin_amdgcn_wmma_f32_16x16x32_bf16(false, f.a0, false, f.b1, (short)0, acc01, false, false);
        acc02 = __builtin_amdgcn_wmma_f32_16x16x32_bf16(false, f.a0, false, f.b2, (short)0, acc02, false, false);
        acc03 = __builtin_amdgcn_wmma_f32_16x16x32_bf16(false, f.a0, false, f.b3, (short)0, acc03, false, false);
        acc10 = __builtin_amdgcn_wmma_f32_16x16x32_bf16(false, f.a1, false, f.b0, (short)0, acc10, false, false);
        acc11 = __builtin_amdgcn_wmma_f32_16x16x32_bf16(false, f.a1, false, f.b1, (short)0, acc11, false, false);
        acc12 = __builtin_amdgcn_wmma_f32_16x16x32_bf16(false, f.a1, false, f.b2, (short)0, acc12, false, false);
        acc13 = __builtin_amdgcn_wmma_f32_16x16x32_bf16(false, f.a1, false, f.b3, (short)0, acc13, false, false);
    };

    // Software-pipelined, double-buffered K loop (K % 64 == 0).
    Frags f0, f1;
    load_step(f0, 0);
    int k0 = 0;
    for (; k0 + 64 < K; k0 += 64) {
        load_step(f1, k0 + 32);
        mma_step(f0);
        load_step(f0, k0 + 64);
        mma_step(f1);
    }
    load_step(f1, K - 32);
    mma_step(f0);
    mma_step(f1);

    // Epilogue.  C/D layout: VGPR r -> M = r + 8*half, N = lane&15.
#pragma unroll
    for (int mf = 0; mf < 2; ++mf) {
        const int cm0 = tileM + mf * 16 + half * 8;
#pragma unroll
        for (int s = 0; s < 4; ++s) {
            const v8f acc =
                (mf == 0) ? ((s == 0) ? acc00 : (s == 1) ? acc01 : (s == 2) ? acc02 : acc03)
                          : ((s == 0) ? acc10 : (s == 1) ? acc11 : (s == 2) ? acc12 : acc13);
            const int cn = tileN0 + s * 16 + mn;
            const float bv = HAS_BIAS ? bias[cn] : 0.f;
#pragma unroll
            for (int r = 0; r < 8; ++r) {
                float v = acc[r] + bv;
                if (SOFTPLUS) v = (v > 20.f) ? v : logf(1.f + __expf(v));
                const size_t o = (size_t)(cm0 + r) * N + cn;
                if (OUT_BF16) outB[o] = (bf16)v;
                else          outF[o] = v;
            }
        }
    }
}

// ---------------------------------------------------------------------------
// Conversions
// ---------------------------------------------------------------------------
__global__ __launch_bounds__(256) void concat_bf16_kernel(
    const float* __restrict__ t, const float* __restrict__ a,
    const float* __restrict__ v, bf16* __restrict__ cat)
{
    int i = blockIdx.x * 256 + threadIdx.x;
    if (i >= Mrows * 3 * H) return;
    int c = i % (3 * H), row = i / (3 * H);
    const float* src = (c < H) ? t : (c < 2 * H) ? a : v;
    int cc = (c < H) ? c : (c < 2 * H) ? c - H : c - 2 * H;
    cat[i] = (bf16)src[(size_t)row * H + cc];
}

// W[K,N] f32 -> Wt[Npad, Kpad] bf16, zero-padded outside [N,K]
__global__ __launch_bounds__(256) void transpose_w_bf16_kernel(
    const float* __restrict__ W, bf16* __restrict__ Wt,
    int K, int N, int Kpad, int Npad)
{
    int i = blockIdx.x * 256 + threadIdx.x;
    if (i >= Npad * Kpad) return;
    int n = i / Kpad, k = i % Kpad;
    Wt[i] = (n < N && k < K) ? (bf16)W[(size_t)k * N + n] : (bf16)0.f;
}

// dbl[:, :48] -> dtb[4096, 64] bf16, zero-padded cols 48..63
__global__ __launch_bounds__(256) void slice_dt_bf16_kernel(
    const float* __restrict__ dbl, bf16* __restrict__ dtb)
{
    int i = blockIdx.x * 256 + threadIdx.x;
    if (i >= Mrows * RrkPad) return;
    int c = i & (RrkPad - 1), row = i >> 6;
    dtb[i] = (c < Rrk) ? (bf16)dbl[(size_t)row * NDBLP + c] : (bf16)0.f;
}

// ---------------------------------------------------------------------------
// Dual LayerNorm: in-place LN1 (proj_ln) on fused, then LN2 (blk_ln) -> h bf16
// ---------------------------------------------------------------------------
__device__ __forceinline__ float block_reduce_sum(float v, float* sm)
{
    int t = threadIdx.x;
    sm[t] = v;
    __syncthreads();
    for (int s = 128; s > 0; s >>= 1) {
        if (t < s) sm[t] += sm[t + s];
        __syncthreads();
    }
    float r = sm[0];
    __syncthreads();
    return r;
}

__global__ __launch_bounds__(256) void ln_dual_kernel(
    float* __restrict__ fused, bf16* __restrict__ hb,
    const float* __restrict__ g1, const float* __restrict__ b1,
    const float* __restrict__ g2, const float* __restrict__ b2)
{
    __shared__ float sm[256];
    const int row = blockIdx.x;
    const int t = threadIdx.x;
    const int c0 = t, c1 = t + 256, c2 = t + 512;
    float* rowp = fused + (size_t)row * H;

    float x0 = rowp[c0], x1 = rowp[c1], x2 = rowp[c2];
    float mu = block_reduce_sum(x0 + x1 + x2, sm) * (1.f / H);
    float d0 = x0 - mu, d1 = x1 - mu, d2 = x2 - mu;
    float var = block_reduce_sum(d0 * d0 + d1 * d1 + d2 * d2, sm) * (1.f / H);
    float rs = rsqrtf(var + 1e-5f);
    float y0 = d0 * rs * g1[c0] + b1[c0];
    float y1 = d1 * rs * g1[c1] + b1[c1];
    float y2 = d2 * rs * g1[c2] + b1[c2];

    float mu2 = block_reduce_sum(y0 + y1 + y2, sm) * (1.f / H);
    float e0 = y0 - mu2, e1 = y1 - mu2, e2 = y2 - mu2;
    float var2 = block_reduce_sum(e0 * e0 + e1 * e1 + e2 * e2, sm) * (1.f / H);
    float rs2 = rsqrtf(var2 + 1e-5f);

    rowp[c0] = y0; rowp[c1] = y1; rowp[c2] = y2;   // residual source (post-LN1)
    bf16* hrow = hb + (size_t)row * H;
    hrow[c0] = (bf16)(e0 * rs2 * g2[c0] + b2[c0]);
    hrow[c1] = (bf16)(e1 * rs2 * g2[c1] + b2[c1]);
    hrow[c2] = (bf16)(e2 * rs2 * g2[c2] + b2[c2]);
}

// ---------------------------------------------------------------------------
// Causal depthwise conv (K=4) + SiLU.  x = first DIN cols of xz[M, 2*DIN].
// ---------------------------------------------------------------------------
__global__ __launch_bounds__(256) void conv_silu_kernel(
    const bf16* __restrict__ xz, const float* __restrict__ cw,
    const float* __restrict__ cb, bf16* __restrict__ xc)
{
    int i = blockIdx.x * 256 + threadIdx.x;
    if (i >= Mrows * DIN) return;
    int d = i % DIN;
    int row = i / DIN;
    int l = row & (Lseq - 1);
    float acc = cb[d];
#pragma unroll
    for (int j = 0; j < Kcv; ++j) {
        int ll = l - (Kcv - 1) + j;
        if (ll >= 0)
            acc += (float)xz[(size_t)(row - (Kcv - 1) + j) * (2 * DIN) + d] * cw[d * Kcv + j];
    }
    xc[i] = (bf16)(acc / (1.f + __expf(-acc)));   // SiLU
}

// ---------------------------------------------------------------------------
// Selective scan: per-thread 64-state recurrence; B/C rows staged in LDS.
// ---------------------------------------------------------------------------
__global__ __launch_bounds__(256) void scan_kernel(
    const float* __restrict__ dbl, const float* __restrict__ delta,
    const bf16* __restrict__ xcb, const bf16* __restrict__ xz,
    const float* __restrict__ A_log, const float* __restrict__ Dp,
    bf16* __restrict__ ybf)
{
    const int b = blockIdx.x / (DIN / 256);
    const int d = (blockIdx.x % (DIN / 256)) * 256 + threadIdx.x;

    float a[Nst], h[Nst];
#pragma unroll
    for (int n = 0; n < Nst; ++n) {
        a[n] = -__expf(A_log[(size_t)d * Nst + n]);
        h[n] = 0.f;
    }
    const float Dd = Dp[d];
    __shared__ float Bs[Nst], Cs[Nst];

    for (int l = 0; l < Lseq; ++l) {
        const int row = b * Lseq + l;
        __syncthreads();
        if (threadIdx.x < 2 * Nst) {
            const int t = threadIdx.x;
            float v = dbl[(size_t)row * NDBLP + Rrk + t];
            if (t < Nst) Bs[t] = v; else Cs[t - Nst] = v;
        }
        __syncthreads();

        const float dl = delta[(size_t)row * DIN + d];
        const float u  = (float)xcb[(size_t)row * DIN + d];
        const float du = dl * u;
        float y = 0.f;
#pragma unroll
        for (int n = 0; n < Nst; ++n) {
            h[n] = __expf(dl * a[n]) * h[n] + du * Bs[n];
            y += h[n] * Cs[n];
        }
        const float z = (float)xz[(size_t)row * (2 * DIN) + DIN + d];
        const float out = (y + u * Dd) * (z / (1.f + __expf(-z)));
        ybf[(size_t)row * DIN + d] = (bf16)out;
    }
}

// ---------------------------------------------------------------------------
// out[b,h] = mean_l( fused[b,l,h] + m[b,l,h] )
// ---------------------------------------------------------------------------
__global__ __launch_bounds__(256) void mean_kernel(
    const float* __restrict__ fused, const float* __restrict__ m,
    float* __restrict__ out)
{
    int i = blockIdx.x * 256 + threadIdx.x;
    if (i >= Bsz * H) return;
    int b = i / H, hh = i % H;
    float s = 0.f;
    for (int l = 0; l < Lseq; ++l) {
        size_t o = (size_t)(b * Lseq + l) * H + hh;
        s += fused[o] + m[o];
    }
    out[i] = s * (1.f / Lseq);
}

// ---------------------------------------------------------------------------
// Launch
// ---------------------------------------------------------------------------
extern "C" void kernel_launch(void* const* d_in, const int* in_sizes, int n_in,
                              void* d_out, int out_size, void* d_ws, size_t ws_size,
                              hipStream_t stream)
{
    const float* text      = (const float*)d_in[0];
    const float* audio     = (const float*)d_in[1];
    const float* video     = (const float*)d_in[2];
    const float* proj_w    = (const float*)d_in[3];
    const float* proj_b    = (const float*)d_in[4];
    const float* proj_ln_g = (const float*)d_in[5];
    const float* proj_ln_b = (const float*)d_in[6];
    const float* blk_ln_g  = (const float*)d_in[7];
    const float* blk_ln_b  = (const float*)d_in[8];
    const float* in_proj_w = (const float*)d_in[9];
    const float* conv_w    = (const float*)d_in[10];
    const float* conv_b    = (const float*)d_in[11];
    const float* x_proj_w  = (const float*)d_in[12];
    const float* dt_proj_w = (const float*)d_in[13];
    const float* dt_proj_b = (const float*)d_in[14];
    const float* A_log     = (const float*)d_in[15];
    const float* Dp        = (const float*)d_in[16];
    const float* out_proj_w= (const float*)d_in[17];

    char* ws = (char*)d_ws;
    size_t off = 0;
    auto alloc = [&](size_t bytes) -> char* {
        char* p = ws + off;
        off += (bytes + 255) & ~(size_t)255;
        return p;
    };

    bf16* cat    = (bf16*)alloc((size_t)Mrows * 3 * H * 2);
    bf16* wprojT = (bf16*)alloc((size_t)H * 3 * H * 2);          // [768, 2304]
    float* fused = (float*)alloc((size_t)Mrows * H * 4);
    bf16* hb     = (bf16*)alloc((size_t)Mrows * H * 2);
    bf16* winT   = (bf16*)alloc((size_t)(2 * DIN) * H * 2);      // [3072, 768]
    bf16* xzb    = (bf16*)alloc((size_t)Mrows * 2 * DIN * 2);
    bf16* xcb    = (bf16*)alloc((size_t)Mrows * DIN * 2);
    bf16* wxT    = (bf16*)alloc((size_t)NDBLP * DIN * 2);        // [192, 1536] zero-padded rows
    float* dbl   = (float*)alloc((size_t)Mrows * NDBLP * 4);     // [4096, 192]
    bf16* dtb    = (bf16*)alloc((size_t)Mrows * RrkPad * 2);     // [4096, 64] zero-padded
    bf16* wdtT   = (bf16*)alloc((size_t)DIN * RrkPad * 2);       // [1536, 64] zero-padded
    float* delta = (float*)alloc((size_t)Mrows * DIN * 4);
    bf16* yb     = (bf16*)alloc((size_t)Mrows * DIN * 2);
    bf16* woutT  = (bf16*)alloc((size_t)H * DIN * 2);            // [768, 1536]
    float* mbuf  = (float*)alloc((size_t)Mrows * H * 4);
    (void)ws_size; (void)in_sizes; (void)n_in; (void)out_size;

    auto cdiv = [](int a, int b) { return (a + b - 1) / b; };

    // --- precision conversion / weight transpose (zero-padded where needed) ---
    concat_bf16_kernel<<<cdiv(Mrows * 3 * H, 256), 256, 0, stream>>>(text, audio, video, cat);
    transpose_w_bf16_kernel<<<cdiv(H * 3 * H, 256), 256, 0, stream>>>(proj_w, wprojT, 3 * H, H, 3 * H, H);
    transpose_w_bf16_kernel<<<cdiv(2 * DIN * H, 256), 256, 0, stream>>>(in_proj_w, winT, H, 2 * DIN, H, 2 * DIN);
    transpose_w_bf16_kernel<<<cdiv(NDBLP * DIN, 256), 256, 0, stream>>>(x_proj_w, wxT, DIN, NDBL, DIN, NDBLP);
    transpose_w_bf16_kernel<<<cdiv(DIN * RrkPad, 256), 256, 0, stream>>>(dt_proj_w, wdtT, Rrk, DIN, RrkPad, DIN);
    transpose_w_bf16_kernel<<<cdiv(H * DIN, 256), 256, 0, stream>>>(out_proj_w, woutT, DIN, H, DIN, H);

    // --- fused = concat @ proj_w + b ---
    gemm_wmma_bf16<true, false, false>
        <<<dim3(H / 64, Mrows / 256), 256, 0, stream>>>(
            cat, wprojT, proj_b, fused, nullptr, Mrows, H, 3 * H);

    // --- LN1 in place + LN2 -> h (bf16) ---
    ln_dual_kernel<<<Mrows, 256, 0, stream>>>(fused, hb, proj_ln_g, proj_ln_b,
                                              blk_ln_g, blk_ln_b);

    // --- xz = h @ in_proj_w  (bf16 out) ---
    gemm_wmma_bf16<false, false, true>
        <<<dim3((2 * DIN) / 64, Mrows / 256), 256, 0, stream>>>(
            hb, winT, nullptr, nullptr, xzb, Mrows, 2 * DIN, H);

    // --- causal depthwise conv + SiLU ---
    conv_silu_kernel<<<cdiv(Mrows * DIN, 256), 256, 0, stream>>>(xzb, conv_w, conv_b, xcb);

    // --- dbl = xc @ x_proj_w (f32 out, N padded to 192) ---
    gemm_wmma_bf16<false, false, false>
        <<<dim3(NDBLP / 64, Mrows / 256), 256, 0, stream>>>(
            xcb, wxT, nullptr, dbl, nullptr, Mrows, NDBLP, DIN);

    // --- dt slice -> bf16 (K padded to 64); delta = softplus(dt @ dt_proj_w + b) ---
    slice_dt_bf16_kernel<<<cdiv(Mrows * RrkPad, 256), 256, 0, stream>>>(dbl, dtb);
    gemm_wmma_bf16<true, true, false>
        <<<dim3(DIN / 64, Mrows / 256), 256, 0, stream>>>(
            dtb, wdtT, dt_proj_b, delta, nullptr, Mrows, DIN, RrkPad);

    // --- selective scan + skip + gate -> y (bf16) ---
    scan_kernel<<<Bsz * (DIN / 256), 256, 0, stream>>>(
        dbl, delta, xcb, xzb, A_log, Dp, yb);

    // --- m = y @ out_proj_w ---
    gemm_wmma_bf16<false, false, false>
        <<<dim3(H / 64, Mrows / 256), 256, 0, stream>>>(
            yb, woutT, nullptr, mbuf, nullptr, Mrows, H, DIN);

    // --- out = mean_l(fused + m) ---
    mean_kernel<<<cdiv(Bsz * H, 256), 256, 0, stream>>>(fused, mbuf, (float*)d_out);
}
